// Head_17248588661548
// MI455X (gfx1250) — matrix operand order repaired and checked by reading, not compiled, not run on hardware
//
#include <hip/hip_runtime.h>
#include <hip/hip_bf16.h>

typedef __attribute__((ext_vector_type(16))) __bf16 v16bf;
typedef __attribute__((ext_vector_type(8)))  __bf16 v8bf;
typedef __attribute__((ext_vector_type(8)))  float  v8f;
typedef __attribute__((ext_vector_type(4)))  float  v4f;

#define HD    64
#define CDIM  1024
#define TLEN  2048
#define BATCH 8

static __device__ __forceinline__ v8f wmma_bf16(v16bf a, v16bf b, v8f c) {
  // (neg_a, A, neg_b, B, c_mod, C, reuse_a, reuse_b)
  return __builtin_amdgcn_wmma_f32_16x16x32_bf16(false, a, false, b, (short)0, c,
                                                 false, false);
}

// ---------------- Phase 0: weights fp32 -> bf16 (L2-resident) ----------------
__global__ __launch_bounds__(256) void cvt_weights(const float* __restrict__ wq,
                                                   const float* __restrict__ wk,
                                                   const float* __restrict__ wv,
                                                   __bf16* __restrict__ wb) {
  int i = blockIdx.x * 256 + threadIdx.x;
  const int N = HD * CDIM;
  if (i < N) {
    wb[i]         = (__bf16)wq[i];
    wb[N + i]     = (__bf16)wk[i];
    wb[2 * N + i] = (__bf16)wv[i];
  }
}

// ---------------- Phase 1: fused Q + (K,V) projection (bf16 WMMA) ------------
// One wave per 16-row tile. Units 0..1023: Q from `index`.
// Units 1024..2047: K AND V fused from one read of `memory` (each input row
// fetched from HBM exactly once -> 134 MB total HBM traffic, the roofline).
// Q,K stored row-major [B*T][64]; V stored transposed [B][64][T].
__global__ __launch_bounds__(256) void proj_qkv(const float* __restrict__ xq,
                                                const float* __restrict__ xm,
                                                const __bf16* __restrict__ wb,
                                                __bf16* __restrict__ qout,
                                                __bf16* __restrict__ kout,
                                                __bf16* __restrict__ vtout) {
  const int wid  = threadIdx.x >> 5;
  const int lane = threadIdx.x & 31;
  const int m = lane & 15, g = lane >> 4;

  const int  u     = blockIdx.x * 8 + wid;   // 0..2047
  const int  which = u >> 10;                // 0 = q, 1 = fused k+v
  const int  tile  = u & 1023;
  const long row0  = (long)tile * 16;        // flat row in [0, B*T)

  const float* X    = (which == 0) ? xq : xm;
  const float* arow = X + (row0 + m) * CDIM;

  if (which == 0) {
    const __bf16* W = wb;  // Wq
    v8f acc0 = {}, acc1 = {}, acc2 = {}, acc3 = {};
    for (int k0 = 0; k0 < CDIM; k0 += 32) {
      v4f f0 = *(const v4f*)(arow + k0 + 16 * g);
      v4f f1 = *(const v4f*)(arow + k0 + 16 * g + 4);
      v4f f2 = *(const v4f*)(arow + k0 + 16 * g + 8);
      v4f f3 = *(const v4f*)(arow + k0 + 16 * g + 12);
      v16bf a;
#pragma unroll
      for (int i = 0; i < 4; ++i) {
        a[i]      = (__bf16)f0[i];
        a[4 + i]  = (__bf16)f1[i];
        a[8 + i]  = (__bf16)f2[i];
        a[12 + i] = (__bf16)f3[i];
      }
      const __bf16* wr = W + k0 + 16 * g;
      v16bf b0 = *(const v16bf*)(wr + (0 * 16 + m) * CDIM);
      v16bf b1 = *(const v16bf*)(wr + (1 * 16 + m) * CDIM);
      v16bf b2 = *(const v16bf*)(wr + (2 * 16 + m) * CDIM);
      v16bf b3 = *(const v16bf*)(wr + (3 * 16 + m) * CDIM);
      acc0 = wmma_bf16(a, b0, acc0);
      acc1 = wmma_bf16(a, b1, acc1);
      acc2 = wmma_bf16(a, b2, acc2);
      acc3 = wmma_bf16(a, b3, acc3);
    }
    __bf16* out = qout + row0 * HD;
#pragma unroll
    for (int r = 0; r < 8; ++r) {
      __bf16* orow = out + (8 * g + r) * HD;
      orow[0  + m] = (__bf16)acc0[r];
      orow[16 + m] = (__bf16)acc1[r];
      orow[32 + m] = (__bf16)acc2[r];
      orow[48 + m] = (__bf16)acc3[r];
    }
  } else {
    const __bf16* Wk2 = wb + (long)1 * (HD * CDIM);
    const __bf16* Wv2 = wb + (long)2 * (HD * CDIM);
    v8f ak0 = {}, ak1 = {}, ak2 = {}, ak3 = {};
    v8f av0 = {}, av1 = {}, av2 = {}, av3 = {};
    for (int k0 = 0; k0 < CDIM; k0 += 32) {
      v4f f0 = *(const v4f*)(arow + k0 + 16 * g);
      v4f f1 = *(const v4f*)(arow + k0 + 16 * g + 4);
      v4f f2 = *(const v4f*)(arow + k0 + 16 * g + 8);
      v4f f3 = *(const v4f*)(arow + k0 + 16 * g + 12);
      v16bf a;
#pragma unroll
      for (int i = 0; i < 4; ++i) {
        a[i]      = (__bf16)f0[i];
        a[4 + i]  = (__bf16)f1[i];
        a[8 + i]  = (__bf16)f2[i];
        a[12 + i] = (__bf16)f3[i];
      }
      const long wrow = (long)m * CDIM + k0 + 16 * g;
      v16bf bk0 = *(const v16bf*)(Wk2 + wrow + (long)(0 * 16) * CDIM);
      v16bf bk1 = *(const v16bf*)(Wk2 + wrow + (long)(1 * 16) * CDIM);
      v16bf bk2 = *(const v16bf*)(Wk2 + wrow + (long)(2 * 16) * CDIM);
      v16bf bk3 = *(const v16bf*)(Wk2 + wrow + (long)(3 * 16) * CDIM);
      ak0 = wmma_bf16(a, bk0, ak0);
      ak1 = wmma_bf16(a, bk1, ak1);
      ak2 = wmma_bf16(a, bk2, ak2);
      ak3 = wmma_bf16(a, bk3, ak3);
      v16bf bv0 = *(const v16bf*)(Wv2 + wrow + (long)(0 * 16) * CDIM);
      v16bf bv1 = *(const v16bf*)(Wv2 + wrow + (long)(1 * 16) * CDIM);
      v16bf bv2 = *(const v16bf*)(Wv2 + wrow + (long)(2 * 16) * CDIM);
      v16bf bv3 = *(const v16bf*)(Wv2 + wrow + (long)(3 * 16) * CDIM);
      av0 = wmma_bf16(a, bv0, av0);
      av1 = wmma_bf16(a, bv1, av1);
      av2 = wmma_bf16(a, bv2, av2);
      av3 = wmma_bf16(a, bv3, av3);
    }
    // K: row-major
    __bf16* out = kout + row0 * HD;
#pragma unroll
    for (int r = 0; r < 8; ++r) {
      __bf16* orow = out + (8 * g + r) * HD;
      orow[0  + m] = (__bf16)ak0[r];
      orow[16 + m] = (__bf16)ak1[r];
      orow[32 + m] = (__bf16)ak2[r];
      orow[48 + m] = (__bf16)ak3[r];
    }
    // V: transposed vt[b][h][t]
    const int b    = (int)(row0 >> 11);
    const int tloc = (int)(row0 & (TLEN - 1));
    __bf16* vbp = vtout + (long)b * HD * TLEN;
#pragma unroll
    for (int r = 0; r < 8; ++r) {
      const int t = tloc + 8 * g + r;
      vbp[(0  + m) * TLEN + t] = (__bf16)av0[r];
      vbp[(16 + m) * TLEN + t] = (__bf16)av1[r];
      vbp[(32 + m) * TLEN + t] = (__bf16)av2[r];
      vbp[(48 + m) * TLEN + t] = (__bf16)av3[r];
    }
  }
}

// ---------------- Phase 2: causal flash attention (bf16 WMMA) ----------------
// One wave per 16-query tile; key tiles of 32 so P is exactly one A-fragment.
__global__ __launch_bounds__(128) void attn(const __bf16* __restrict__ q,
                                            const __bf16* __restrict__ k,
                                            const __bf16* __restrict__ vt,
                                            float* __restrict__ out) {
  __shared__ __align__(64) __bf16 plds_all[4 * 16 * 32];
  const int wid  = threadIdx.x >> 5;
  const int lane = threadIdx.x & 31;
  const int m = lane & 15, g = lane >> 4;
  __bf16* plds = plds_all + wid * (16 * 32);

  const int qt = blockIdx.x * 4 + wid;   // 0..1023
  const int b  = qt >> 7;                // 128 q-tiles per batch
  const int t0 = (qt & 127) * 16;

  const __bf16* qb  = q  + ((long)b * TLEN + t0) * HD;
  const __bf16* kb  = k  + (long)b * TLEN * HD;
  const __bf16* vtb = vt + (long)b * HD * TLEN;
  float*        ob  = out + ((long)b * TLEN + t0) * HD;

  // Q A-fragments for the two 32-wide K-steps over H=64
  v16bf qa0, qa1;
  {
    const __bf16* qr = qb + m * HD;
    v8bf lo = *(const v8bf*)(qr + 16 * g);
    v8bf hi = *(const v8bf*)(qr + 8 + 16 * g);
#pragma unroll
    for (int i = 0; i < 8; ++i) { qa0[i] = lo[i]; qa0[8 + i] = hi[i]; }
    lo = *(const v8bf*)(qr + 32 + 16 * g);
    hi = *(const v8bf*)(qr + 40 + 16 * g);
#pragma unroll
    for (int i = 0; i < 8; ++i) { qa1[i] = lo[i]; qa1[8 + i] = hi[i]; }
  }

  float mI[8], lI[8];
#pragma unroll
  for (int r = 0; r < 8; ++r) { mI[r] = -__builtin_inff(); lI[r] = 0.f; }
  v8f o0 = {}, o1 = {}, o2 = {}, o3 = {};
  const float scale = 0.03125f;  // 1/sqrt(C) = 1/32

  for (int s0 = 0; s0 <= t0 + 15; s0 += 32) {
    // ---- scores S = Q @ K^T for keys [s0, s0+32) as two 16x16 D-frags ----
    v8f sa = {}, sb = {};
    {
      const __bf16* kra = kb + (s0 + m) * HD;        // keys s0+n   (n = m)
      const __bf16* krb = kb + (s0 + 16 + m) * HD;   // keys s0+16+n
      v16bf ba0 = *(const v16bf*)(kra + 16 * g);
      v16bf ba1 = *(const v16bf*)(kra + 32 + 16 * g);
      v16bf bb0 = *(const v16bf*)(krb + 16 * g);
      v16bf bb1 = *(const v16bf*)(krb + 32 + 16 * g);
      sa = wmma_bf16(qa0, ba0, sa);
      sa = wmma_bf16(qa1, ba1, sa);
      sb = wmma_bf16(qa0, bb0, sb);
      sb = wmma_bf16(qa1, bb1, sb);
    }
    if (s0 + 32 < TLEN)
      __builtin_prefetch((const void*)(kb + (s0 + 32 + m) * HD), 0, 0);

    // ---- scale + causal mask (wave-uniform branch) ----
    float pa[8], pb[8], alpha[8];
    const bool need_mask = (s0 + 31 > t0);
#pragma unroll
    for (int r = 0; r < 8; ++r) {
      float va = sa[r] * scale;
      float vb = sb[r] * scale;
      if (need_mask) {
        const int t = t0 + 8 * g + r;
        if (s0 + m      > t) va = -__builtin_inff();
        if (s0 + 16 + m > t) vb = -__builtin_inff();
      }
      pa[r] = va; pb[r] = vb;
    }

    // ---- online softmax: rows live across 16-lane halves -> width-16 shfls --
#pragma unroll
    for (int r = 0; r < 8; ++r) {
      float v = fmaxf(pa[r], pb[r]);
      for (int off = 8; off > 0; off >>= 1) v = fmaxf(v, __shfl_xor(v, off, 16));
      const float mNew = fmaxf(mI[r], v);
      alpha[r] = __expf(mI[r] - mNew);
      mI[r] = mNew;
      pa[r] = __expf(pa[r] - mNew);
      pb[r] = __expf(pb[r] - mNew);
      float rs = pa[r] + pb[r];
      for (int off = 8; off > 0; off >>= 1) rs += __shfl_xor(rs, off, 16);
      lI[r] = lI[r] * alpha[r] + rs;
    }

    // ---- D-layout -> A-layout transpose of P through LDS ----
#pragma unroll
    for (int r = 0; r < 8; ++r) {
      plds[(8 * g + r) * 32 + m]      = (__bf16)pa[r];
      plds[(8 * g + r) * 32 + 16 + m] = (__bf16)pb[r];
    }
    __asm__ volatile("s_wait_dscnt 0" ::: "memory");  // same-wave LDS RAW
    v16bf pfrag;
    {
      v8bf lo = *(const v8bf*)(plds + m * 32 + 16 * g);
      v8bf hi = *(const v8bf*)(plds + m * 32 + 8 + 16 * g);
#pragma unroll
      for (int i = 0; i < 8; ++i) { pfrag[i] = lo[i]; pfrag[8 + i] = hi[i]; }
    }

    // ---- rescale O, then O += P @ V  (B from transposed V: contiguous) ----
#pragma unroll
    for (int r = 0; r < 8; ++r) {
      o0[r] *= alpha[r]; o1[r] *= alpha[r];
      o2[r] *= alpha[r]; o3[r] *= alpha[r];
    }
    {
      const __bf16* vr = vtb + m * TLEN + s0 + 16 * g;   // row h0+m of V^T
      v16bf vb0 = *(const v16bf*)(vr + 0 * 16 * TLEN);
      v16bf vb1 = *(const v16bf*)(vr + 1 * 16 * TLEN);
      v16bf vb2 = *(const v16bf*)(vr + 2 * 16 * TLEN);
      v16bf vb3 = *(const v16bf*)(vr + 3 * 16 * TLEN);
      o0 = wmma_bf16(pfrag, vb0, o0);
      o1 = wmma_bf16(pfrag, vb1, o1);
      o2 = wmma_bf16(pfrag, vb2, o2);
      o3 = wmma_bf16(pfrag, vb3, o3);
    }
  }

  // ---- epilogue: normalize and store fp32 ----
#pragma unroll
  for (int r = 0; r < 8; ++r) {
    const float inv = 1.f / lI[r];
    float* orow = ob + (8 * g + r) * HD;
    orow[0  + m] = o0[r] * inv;
    orow[16 + m] = o1[r] * inv;
    orow[32 + m] = o2[r] * inv;
    orow[48 + m] = o3[r] * inv;
  }
}

// ---------------------------------------------------------------------------
extern "C" void kernel_launch(void* const* d_in, const int* in_sizes, int n_in,
                              void* d_out, int out_size, void* d_ws, size_t ws_size,
                              hipStream_t stream) {
  const float* index  = (const float*)d_in[0];  // [B,T,C]
  const float* memory = (const float*)d_in[1];  // [B,T,C]
  const float* Wq     = (const float*)d_in[2];  // [H,C]
  const float* Wk     = (const float*)d_in[3];
  const float* Wv     = (const float*)d_in[4];
  float* out = (float*)d_out;                   // [B,T,H] fp32

  char* ws = (char*)d_ws;
  __bf16* wb  = (__bf16*)ws;                                    // 3*H*C bf16
  __bf16* qws = (__bf16*)(ws + (size_t)3 * HD * CDIM * 2);      // [B*T][H]
  __bf16* kws = qws + (size_t)BATCH * TLEN * HD;                // [B*T][H]
  __bf16* vws = kws + (size_t)BATCH * TLEN * HD;                // [B][H][T]

  cvt_weights<<<(HD * CDIM + 255) / 256, 256, 0, stream>>>(Wq, Wk, Wv, wb);
  proj_qkv<<<256, 256, 0, stream>>>(index, memory, wb, qws, kws, vws);
  attn<<<256, 128, 0, stream>>>(qws, kws, vws, out);
}
